// CBoW_3221225472040
// MI455X (gfx1250) — compile-verified, matrix operand order checked by reading, not run on hardware
//
#include <hip/hip_runtime.h>
#include <hip/hip_bf16.h>
#include <hip/hip_fp16.h>

typedef __attribute__((ext_vector_type(16))) _Float16 v16h;
typedef __attribute__((ext_vector_type(8)))  _Float16 v8h;
typedef __attribute__((ext_vector_type(8)))  float    v8f;

#define SEQ   200
#define BATCH 1024
#define DIM   300
#define HID   600
#define KPAD  640   // K padded to multiple of 32 (WMMA K) and 64B rows
#define NPAD  608   // N padded to multiple of 16

// ---------------------------------------------------------------------------
// Kernel 1: per-batch-column dedup + embedding-bag sum, emitted as f16 rows
// of the padded A matrix [BATCH, KPAD]. cols 0..299 = e_learn, 300..599 =
// e_static, 600..639 = zero pad.
// ---------------------------------------------------------------------------
__global__ void __launch_bounds__(320)
embed_kernel(const int* __restrict__ tokens,
             const float* __restrict__ lut,
             const float* __restrict__ slut,
             _Float16* __restrict__ A) {
  __shared__ int toks[SEQ];
  __shared__ int list[SEQ];
  __shared__ unsigned char keep[SEQ];
  __shared__ int cnt;

  const int b   = blockIdx.x;
  const int tid = threadIdx.x;

  if (tid < SEQ) toks[tid] = tokens[tid * BATCH + b];
  __syncthreads();

  if (tid < SEQ) {
    const int t = toks[tid];
    int k = 1;
    for (int j = 0; j < tid; ++j) {
      if (toks[j] == t) { k = 0; break; }
    }
    keep[tid] = (unsigned char)k;
  }
  __syncthreads();

  if (tid == 0) {
    int c = 0;
    for (int s = 0; s < SEQ; ++s)
      if (keep[s]) list[c++] = toks[s];
    cnt = c;
  }
  __syncthreads();

  const int n = cnt;
  _Float16* arow = A + (size_t)b * KPAD;
  if (tid < DIM) {
    float al = 0.f, as = 0.f;
    for (int i = 0; i < n; ++i) {
      const size_t t = (size_t)list[i];
      al += lut [t * DIM + tid];   // 1200B coalesced burst across lanes
      as += slut[t * DIM + tid];
    }
    arow[tid]       = (_Float16)al;
    arow[DIM + tid] = (_Float16)as;
  } else if (tid < DIM + 20) {
    const int p = HID + (tid - DIM) * 2;  // zero-pad cols 600..639
    arow[p]     = (_Float16)0.f;
    arow[p + 1] = (_Float16)0.f;
  }
}

// ---------------------------------------------------------------------------
// Kernel 2: W1 [600,600] f32 -> B f16 [NPAD, KPAD], zero padded.
// Row-major W1 == B-matrix source since B[k][n] = W1[n][k] and the WMMA
// B fragment indexes lane = N, vgpr/half = K.
// ---------------------------------------------------------------------------
__global__ void w1_cvt_kernel(const float* __restrict__ W1,
                              _Float16* __restrict__ B) {
  const int idx = blockIdx.x * blockDim.x + threadIdx.x;
  if (idx >= NPAD * KPAD) return;
  const int n = idx / KPAD;
  const int k = idx % KPAD;
  const float v = (n < HID && k < HID) ? W1[n * HID + k] : 0.f;
  B[idx] = (_Float16)v;
}

// ---------------------------------------------------------------------------
// Kernel 3: H = relu(A x W1^T + b1) via v_wmma_f32_16x16x32_f16.
// One wave per 16x16 output tile; K loop of KPAD/32 = 20 WMMAs.
// Per the CDNA5 16-bit fragment layout, each lane's v16h is two contiguous
// 8-half (16B) chunks: lanes 0-15 -> K {0..7, 16..23}, lanes 16-31 ->
// K {8..15, 24..31}.
// ---------------------------------------------------------------------------
__global__ void __launch_bounds__(32)
gemm_wmma_kernel(const _Float16* __restrict__ A,
                 const _Float16* __restrict__ B,
                 const float* __restrict__ b1,
                 float* __restrict__ H) {
  const int mt   = blockIdx.x;      // 0..63  (batch tiles)
  const int nt   = blockIdx.y;      // 0..37  (output-feature tiles)
  const int lane = threadIdx.x;     // 0..31
  const int hsel = lane >> 4;       // 0 or 1
  const int r    = lane & 15;

  const _Float16* arow = A + (size_t)(mt * 16 + r) * KPAD + hsel * 8;
  const _Float16* brow = B + (size_t)(nt * 16 + r) * KPAD + hsel * 8;

  v8f c = {};
  for (int ks = 0; ks < KPAD / 32; ++ks) {
    const int k0 = ks * 32;
    v16h a, bm;
    *(v8h*)&a        = *(const v8h*)(arow + k0);       // global_load_b128
    *((v8h*)&a + 1)  = *(const v8h*)(arow + k0 + 16);
    *(v8h*)&bm       = *(const v8h*)(brow + k0);
    *((v8h*)&bm + 1) = *(const v8h*)(brow + k0 + 16);
    c = __builtin_amdgcn_wmma_f32_16x16x32_f16(
        /*neg_a=*/false, a, /*neg_b=*/false, bm,
        /*c_mod=*/(short)0, c, /*reuse_a=*/false, /*reuse_b=*/false);
  }

  const int col = nt * 16 + r;
  if (col < HID) {
    const float bias = b1[col];
    const int mbase = mt * 16 + ((lane < 16) ? 0 : 8);  // C/D layout: vgpr i -> M=i / M=8+i
#pragma unroll
    for (int i = 0; i < 8; ++i) {
      float v = c[i] + bias;
      H[(size_t)(mbase + i) * HID + col] = v > 0.f ? v : 0.f;
    }
  }
}

// ---------------------------------------------------------------------------
// Kernel 4: out[b] = H[b,:] . W2 + b2, one wave32 per row.
// ---------------------------------------------------------------------------
__global__ void __launch_bounds__(256)
out_kernel(const float* __restrict__ H,
           const float* __restrict__ W2,
           const float* __restrict__ b2,
           float* __restrict__ out) {
  const int wave = threadIdx.x >> 5;
  const int lane = threadIdx.x & 31;
  const int row  = blockIdx.x * (blockDim.x >> 5) + wave;
  if (row >= BATCH) return;

  const float* h = H + (size_t)row * HID;
  float s = 0.f;
  for (int n = lane; n < HID; n += 32) s += h[n] * W2[n];
#pragma unroll
  for (int off = 16; off > 0; off >>= 1) s += __shfl_xor(s, off, 32);
  if (lane == 0) out[row] = s + b2[0];
}

// ---------------------------------------------------------------------------
extern "C" void kernel_launch(void* const* d_in, const int* in_sizes, int n_in,
                              void* d_out, int out_size, void* d_ws, size_t ws_size,
                              hipStream_t stream) {
  const int*   tokens = (const int*)  d_in[0];
  const float* lut    = (const float*)d_in[1];
  const float* slut   = (const float*)d_in[2];
  const float* W1     = (const float*)d_in[3];
  const float* b1     = (const float*)d_in[4];
  const float* W2     = (const float*)d_in[5];
  const float* b2     = (const float*)d_in[6];
  float* out = (float*)d_out;

  char* ws = (char*)d_ws;
  _Float16* A = (_Float16*)ws;                                   // 1024*640*2 B
  _Float16* B = (_Float16*)(ws + (size_t)BATCH * KPAD * 2);      //  608*640*2 B
  float*    H = (float*)   (ws + (size_t)BATCH * KPAD * 2
                               + (size_t)NPAD  * KPAD * 2);      // 1024*600*4 B

  embed_kernel<<<BATCH, 320, 0, stream>>>(tokens, lut, slut, A);
  w1_cvt_kernel<<<(NPAD * KPAD + 255) / 256, 256, 0, stream>>>(W1, B);
  gemm_wmma_kernel<<<dim3(BATCH / 16, NPAD / 16), 32, 0, stream>>>(A, B, b1, H);
  out_kernel<<<BATCH / 8, 256, 0, stream>>>(H, W2, b2, out);
}
